// UnifiedModalEncoder_85564338471298
// MI455X (gfx1250) — matrix operand order repaired and checked by reading, not compiled, not run on hardware
//
#include <hip/hip_runtime.h>
#include <math.h>

// ---------------- problem constants ----------------
#define B_    8
#define S_    512
#define E_    1024
#define H_    16
#define HD_   64
#define T_    (B_ * S_)          // 4096 tokens
#define DFF_  4096
#define NS_   4
#define NM_   2
#define NE_   6
#define EPSLN 1e-5f

typedef __attribute__((ext_vector_type(16))) _Float16 v16h;
typedef __attribute__((ext_vector_type(8)))  _Float16 v8h;
typedef __attribute__((ext_vector_type(4)))  _Float16 v4h;
typedef __attribute__((ext_vector_type(8)))  float    v8f;
typedef __attribute__((ext_vector_type(4)))  float    v4f;

__device__ __forceinline__ float gelu_exact(float x) {
    return 0.5f * x * (1.0f + erff(x * 0.70710678118654752440f));
}

// ================= fp32 -> f16 conversion (weights, once per launch) =================
__global__ __launch_bounds__(256)
void cvt_kernel(const float* __restrict__ src, _Float16* __restrict__ dst, int n4) {
    int i = blockIdx.x * blockDim.x + threadIdx.x;
    if (i >= n4) return;
    v4f v = ((const v4f*)src)[i];
    v4h h;
    #pragma unroll
    for (int q = 0; q < 4; ++q) h[q] = (_Float16)v[q];
    ((v4h*)dst)[i] = h;
}

// ================= LayerNorm: one wave32 per token; fp32 and/or f16 out =================
__global__ __launch_bounds__(256)
void ln_kernel(const float* __restrict__ x, const float* __restrict__ w,
               const float* __restrict__ b, float* __restrict__ outF,
               _Float16* __restrict__ outH) {
    int warp = (blockIdx.x * blockDim.x + threadIdx.x) >> 5;
    int lane = threadIdx.x & 31;
    if (warp >= T_) return;
    const float* xr = x + (size_t)warp * E_;
    float s = 0.f, ss = 0.f;
    for (int i = lane; i < E_; i += 32) { float v = xr[i]; s += v; ss += v * v; }
    #pragma unroll
    for (int m = 16; m; m >>= 1) { s += __shfl_xor(s, m, 32); ss += __shfl_xor(ss, m, 32); }
    float mean = s * (1.0f / E_);
    float var  = ss * (1.0f / E_) - mean * mean;
    float inv  = rsqrtf(var + EPSLN);
    for (int i = lane; i < E_; i += 32) {
        float v = (xr[i] - mean) * inv * w[i] + b[i];
        if (outF) outF[(size_t)warp * E_ + i] = v;
        outH[(size_t)warp * E_ + i] = (_Float16)v;
    }
}

// ================= WMMA GEMM, f16 operands, async-to-LDS double buffered =================
// C[M,N] = epilogue( A[M,Kd](f16) * W[N,Kd](f16)^T + bias[N] )
// Block tile 256x128, 8 waves as 4(M) x 2(N), wave tile 64x64 -> 16 WMMA : 16 ds_load_b128.
#define BM  256
#define BN  128
#define BK  32
#define LDK 40      // padded LDS row (halves): 80B stride, keeps 16B-aligned b128 chunks

enum { EP_BIAS = 0, EP_GELU = 1, EP_MASK = 2, EP_RES = 3 };
// EP_BIAS, EP_GELU write f16; EP_MASK, EP_RES write fp32.

template <int EP>
__global__ __launch_bounds__(256)
void gemm_wmma(const _Float16* __restrict__ A, const _Float16* __restrict__ W,
               const float* __restrict__ bias, void* __restrict__ Cv,
               int M, int N, int Kd,
               const float* __restrict__ mask, int maskLd, int maskCol,
               const float* __restrict__ res) {
    __shared__ alignas(16) _Float16 As[2][BM][LDK];
    __shared__ alignas(16) _Float16 Bs[2][BN][LDK];

    const int tid  = threadIdx.x;
    const int m0   = blockIdx.y * BM;
    const int n0   = blockIdx.x * BN;
    const int wave = tid >> 5;
    const int lane = tid & 31;
    const int r    = lane & 15;
    const int hi   = lane >> 4;
    const int wm   = (wave & 3) * 64;      // wave tile 64(M) x 64(N)
    const int wn   = (wave >> 2) * 64;

    // async staging: 16B chunks; A tile = 1024 chunks (4/thread), B tile = 512 (2/thread)
    const unsigned ldsA0  = (unsigned)(uintptr_t)(&As[0][0][0]);
    const unsigned ldsB0  = (unsigned)(uintptr_t)(&Bs[0][0][0]);
    const unsigned bufSzA = (unsigned)(BM * LDK * sizeof(_Float16));
    const unsigned bufSzB = (unsigned)(BN * LDK * sizeof(_Float16));

    auto issue = [&](int k0, int buf) {
        const unsigned boA = (unsigned)buf * bufSzA;
        const unsigned boB = (unsigned)buf * bufSzB;
        #pragma unroll
        for (int t = 0; t < 4; ++t) {
            int c   = tid + t * 256;
            int row = c >> 2;
            int seg = (c & 3) * 8;                       // halves
            unsigned loff = (unsigned)(row * LDK + seg) * 2u;
            unsigned long long ga = (unsigned long long)(uintptr_t)
                (A + (size_t)(m0 + row) * Kd + k0 + seg);
            asm volatile("global_load_async_to_lds_b128 %0, %1, off"
                         :: "v"(ldsA0 + boA + loff), "v"(ga) : "memory");
        }
        #pragma unroll
        for (int t = 0; t < 2; ++t) {
            int c   = tid + t * 256;
            int row = c >> 2;
            int seg = (c & 3) * 8;
            unsigned loff = (unsigned)(row * LDK + seg) * 2u;
            unsigned long long gb = (unsigned long long)(uintptr_t)
                (W + (size_t)(n0 + row) * Kd + k0 + seg);
            asm volatile("global_load_async_to_lds_b128 %0, %1, off"
                         :: "v"(ldsB0 + boB + loff), "v"(gb) : "memory");
        }
    };

    v8f acc[4][4] = {};
    const int nk = Kd / BK;

    issue(0, 0);
    for (int i = 0; i < nk; ++i) {
        const int buf = i & 1;
        if (i + 1 < nk) {
            issue((i + 1) * BK, buf ^ 1);
            asm volatile("s_wait_asynccnt 0x6" ::: "memory");   // tile i landed
        } else {
            asm volatile("s_wait_asynccnt 0x0" ::: "memory");
        }
        __syncthreads();

        v16h af[4], bf[4];
        #pragma unroll
        for (int ii = 0; ii < 4; ++ii) {
            // A 16x32 layout: lane-half hi holds K {hi*8..+7, 16+hi*8..+7}
            const _Float16* pA = &As[buf][wm + ii * 16 + r][hi * 8];
            v8h lo = *(const v8h*)pA;
            v8h h8 = *(const v8h*)(pA + 16);
            af[ii] = __builtin_shufflevector(lo, h8, 0,1,2,3,4,5,6,7,8,9,10,11,12,13,14,15);
        }
        #pragma unroll
        for (int j = 0; j < 4; ++j) {
            // B 32x16 layout: lane-half hi holds K {hi*8..+7, 16+hi*8..+7}
            const _Float16* pB = &Bs[buf][wn + j * 16 + r][hi * 8];
            v8h lo = *(const v8h*)pB;
            v8h h8 = *(const v8h*)(pB + 16);
            bf[j] = __builtin_shufflevector(lo, h8, 0,1,2,3,4,5,6,7,8,9,10,11,12,13,14,15);
        }
        #pragma unroll
        for (int ii = 0; ii < 4; ++ii)
            #pragma unroll
            for (int j = 0; j < 4; ++j)
                acc[ii][j] = __builtin_amdgcn_wmma_f32_16x16x32_f16(
                    false, af[ii], false, bf[j], (short)0, acc[ii][j], false, false);
        __syncthreads();
    }

    // ---- epilogue: D layout -> row = v + hi*8, col = r ----
    #pragma unroll
    for (int ii = 0; ii < 4; ++ii) {
        #pragma unroll
        for (int j = 0; j < 4; ++j) {
            int col = n0 + wn + j * 16 + r;
            float bv = bias[col];
            #pragma unroll
            for (int v = 0; v < 8; ++v) {
                int row = m0 + wm + ii * 16 + hi * 8 + v;
                size_t idx = (size_t)row * N + col;
                float val = acc[ii][j][v] + bv;
                if (EP == EP_BIAS)      ((_Float16*)Cv)[idx] = (_Float16)val;
                else if (EP == EP_GELU) ((_Float16*)Cv)[idx] = (_Float16)gelu_exact(val);
                else if (EP == EP_MASK) ((float*)Cv)[idx] += mask[(size_t)row * maskLd + maskCol] * val;
                else                    ((float*)Cv)[idx] = res[idx] + val;
            }
        }
    }
}

// ================= Attention: one block per (b,q), warp per head (f16 qkv) =================
__global__ __launch_bounds__(512)
void attn_kernel(const _Float16* __restrict__ qkv, _Float16* __restrict__ ctx,
                 float* __restrict__ attnw) {
    __shared__ float Qs[H_][HD_];
    __shared__ float Ps[H_][S_];
    const int b = blockIdx.x / S_;
    const int q = blockIdx.x % S_;
    const int h = threadIdx.x >> 5;
    const int lane = threadIdx.x & 31;
    const size_t rowq = (size_t)(b * S_ + q) * (3 * E_);

    Qs[h][lane]      = (float)qkv[rowq + h * HD_ + lane];
    Qs[h][lane + 32] = (float)qkv[rowq + h * HD_ + lane + 32];
    __syncthreads();

    for (int k = lane; k < S_; k += 32) {
        const v8h* kp = (const v8h*)(qkv + (size_t)(b * S_ + k) * (3 * E_) + E_ + h * HD_);
        float s = 0.f;
        #pragma unroll
        for (int d8 = 0; d8 < HD_ / 8; ++d8) {
            v8h kv = kp[d8];
            #pragma unroll
            for (int u = 0; u < 8; ++u) s += Qs[h][d8 * 8 + u] * (float)kv[u];
        }
        Ps[h][k] = s * 0.125f;   // 1/sqrt(64)
    }
    float mx = -3.0e38f;
    for (int k = lane; k < S_; k += 32) mx = fmaxf(mx, Ps[h][k]);
    #pragma unroll
    for (int m = 16; m; m >>= 1) mx = fmaxf(mx, __shfl_xor(mx, m, 32));
    float sum = 0.f;
    for (int k = lane; k < S_; k += 32) { float e = expf(Ps[h][k] - mx); Ps[h][k] = e; sum += e; }
    #pragma unroll
    for (int m = 16; m; m >>= 1) sum += __shfl_xor(sum, m, 32);
    float inv = 1.0f / sum;
    for (int k = lane; k < S_; k += 32) Ps[h][k] *= inv;

    float c0 = 0.f, c1 = 0.f;
    for (int k = 0; k < S_; ++k) {
        float p = Ps[h][k];
        const _Float16* vp = qkv + (size_t)(b * S_ + k) * (3 * E_) + 2 * E_ + h * HD_;
        c0 += p * (float)vp[lane];
        c1 += p * (float)vp[lane + 32];
    }
    ctx[(size_t)(b * S_ + q) * E_ + h * HD_ + lane]      = (_Float16)c0;
    ctx[(size_t)(b * S_ + q) * E_ + h * HD_ + lane + 32] = (_Float16)c1;

    __syncthreads();
    int k = threadIdx.x;   // 512 threads == S_
    float a = 0.f;
    #pragma unroll
    for (int hh = 0; hh < H_; ++hh) a += Ps[hh][k];
    attnw[(size_t)(b * S_ + q) * S_ + k] = a * (1.0f / H_);
}

// ================= Router: one wave per token (fp32 inputs) =================
template <int NEXP>
__global__ __launch_bounds__(256)
void router_kernel(const float* __restrict__ xn, const float* __restrict__ rw,
                   const float* __restrict__ rb, float* __restrict__ maskOut,
                   float* __restrict__ impAcc, float* __restrict__ loadAcc) {
    int warp = (blockIdx.x * blockDim.x + threadIdx.x) >> 5;
    int lane = threadIdx.x & 31;
    if (warp >= T_) return;
    const float* xr = xn + (size_t)warp * E_;
    float lg[NEXP];
    #pragma unroll
    for (int n = 0; n < NEXP; ++n) {
        float s = 0.f;
        for (int e = lane; e < E_; e += 32) s += xr[e] * rw[n * E_ + e];
        #pragma unroll
        for (int m = 16; m; m >>= 1) s += __shfl_xor(s, m, 32);
        lg[n] = s + rb[n];
    }
    float mx = lg[0];
    #pragma unroll
    for (int n = 1; n < NEXP; ++n) mx = fmaxf(mx, lg[n]);
    float sum = 0.f;
    #pragma unroll
    for (int n = 0; n < NEXP; ++n) { lg[n] = expf(lg[n] - mx); sum += lg[n]; }
    float inv = 1.0f / sum;
    #pragma unroll
    for (int n = 0; n < NEXP; ++n) lg[n] *= inv;
    int i1 = 0;
    #pragma unroll
    for (int n = 1; n < NEXP; ++n) if (lg[n] > lg[i1]) i1 = n;
    int i2 = (i1 == 0) ? 1 : 0;
    #pragma unroll
    for (int n = 0; n < NEXP; ++n) if (n != i1 && n != i2 && lg[n] > lg[i2]) i2 = n;
    if (lane == 0) {
        #pragma unroll
        for (int n = 0; n < NEXP; ++n) {
            bool sel = (n == i1) || (n == i2);
            maskOut[(size_t)warp * NEXP + n] = sel ? lg[n] : 0.f;
            atomicAdd(&impAcc[n], lg[n]);
            if (sel) atomicAdd(&loadAcc[n], 1.0f);
        }
    }
}

// ================= misc =================
__global__ void copy_kernel(const float* __restrict__ a, float* __restrict__ b, int n) {
    int i = blockIdx.x * blockDim.x + threadIdx.x;
    if (i < n) b[i] = a[i];
}
__global__ void zero_kernel(float* __restrict__ p, int n) {
    int i = blockIdx.x * blockDim.x + threadIdx.x;
    if (i < n) p[i] = 0.f;
}
// rs layout: [0..3]=imp_s, [4..7]=load_s, [8..9]=imp_i, [10..11]=load_i (sums)
__global__ void loss_kernel(const float* __restrict__ rs, float* __restrict__ out) {
    if (threadIdx.x == 0 && blockIdx.x == 0) {
        const float invT = 1.0f / (float)T_;
        float l1 = 0.f, l2 = 0.f;
        for (int n = 0; n < NS_; ++n) l1 += (rs[n] * invT) * (rs[NS_ + n] * invT);
        for (int n = 0; n < NM_; ++n) l2 += (rs[8 + n] * invT) * (rs[10 + n] * invT);
        out[0] = (float)NS_ * l1 + (float)NM_ * l2;
    }
}

// ================= host-side orchestration =================
extern "C" void kernel_launch(void* const* d_in, const int* in_sizes, int n_in,
                              void* d_out, int out_size, void* d_ws, size_t ws_size,
                              hipStream_t stream) {
    const float* x         = (const float*)d_in[0];
    const float* norm1_w   = (const float*)d_in[1];
    const float* norm1_b   = (const float*)d_in[2];
    const float* norm2_w   = (const float*)d_in[3];
    const float* norm2_b   = (const float*)d_in[4];
    const float* in_w      = (const float*)d_in[5];
    const float* in_b      = (const float*)d_in[6];
    const float* out_w     = (const float*)d_in[7];
    const float* out_b     = (const float*)d_in[8];
    const float* shared_rw = (const float*)d_in[9];
    const float* shared_rb = (const float*)d_in[10];
    const float* image_rw  = (const float*)d_in[11];
    const float* image_rb  = (const float*)d_in[12];
    const float* exp_w1    = (const float*)d_in[13];
    const float* exp_b1    = (const float*)d_in[14];
    const float* exp_w2    = (const float*)d_in[15];
    const float* exp_b2    = (const float*)d_in[16];
    const float* ffn_w1    = (const float*)d_in[17];
    const float* ffn_b1    = (const float*)d_in[18];
    const float* ffn_w2    = (const float*)d_in[19];
    const float* ffn_b2    = (const float*)d_in[20];

    float* out = (float*)d_out;
    const size_t OX  = 0;
    const size_t OL  = (size_t)T_ * E_;
    const size_t OSM = OL + 1;
    const size_t OIM = OSM + (size_t)T_ * NS_;
    const size_t OAW = OIM + (size_t)T_ * NM_;

    // ---- workspace layout (byte carve-out, 256B aligned) ----
    char* wp = (char*)d_ws;
    auto carve = [&](size_t bytes) {
        char* p = wp;
        wp += (bytes + 255) & ~(size_t)255;
        return (void*)p;
    };
    _Float16* W_INh  = (_Float16*)carve((size_t)3 * E_ * E_ * 2);
    _Float16* W_OUTh = (_Float16*)carve((size_t)E_ * E_ * 2);
    _Float16* W_E1h  = (_Float16*)carve((size_t)NE_ * DFF_ * E_ * 2);
    _Float16* W_E2h  = (_Float16*)carve((size_t)NE_ * E_ * DFF_ * 2);
    _Float16* W_F1h  = (_Float16*)carve((size_t)DFF_ * E_ * 2);
    _Float16* W_F2h  = (_Float16*)carve((size_t)E_ * DFF_ * 2);
    _Float16* XNh    = (_Float16*)carve((size_t)T_ * E_ * 2);       // xn1 / xn3
    _Float16* QKVh   = (_Float16*)carve((size_t)T_ * 3 * E_ * 2);
    _Float16* CTXh   = (_Float16*)carve((size_t)T_ * E_ * 2);
    _Float16* XN2h   = (_Float16*)carve((size_t)T_ * E_ * 2);
    _Float16* HBh    = (_Float16*)carve((size_t)T_ * DFF_ * 2);     // expert/ffn hidden
    float*    X1     = (float*)carve((size_t)T_ * E_ * 4);          // x after attention
    float*    XN2    = (float*)carve((size_t)T_ * E_ * 4);          // fp32 for router
    float*    CMB    = (float*)carve((size_t)T_ * E_ * 4);          // x1 + expert combine
    float*    RS     = (float*)carve(12 * 4);

    const int lnBlocks = (T_ * 32) / 256;
    dim3 blk(256);
    auto cvt = [&](const float* s, _Float16* d, size_t n) {
        int n4 = (int)(n / 4);
        cvt_kernel<<<(n4 + 255) / 256, blk, 0, stream>>>(s, d, n4);
    };

    // 0) weights -> f16 (L2-resident afterwards)
    cvt(in_w,   W_INh,  (size_t)3 * E_ * E_);
    cvt(out_w,  W_OUTh, (size_t)E_ * E_);
    cvt(exp_w1, W_E1h,  (size_t)NE_ * DFF_ * E_);
    cvt(exp_w2, W_E2h,  (size_t)NE_ * E_ * DFF_);
    cvt(ffn_w1, W_F1h,  (size_t)DFF_ * E_);
    cvt(ffn_w2, W_F2h,  (size_t)E_ * DFF_);

    // 1) LN1 (f16 out only)
    ln_kernel<<<lnBlocks, blk, 0, stream>>>(x, norm1_w, norm1_b, nullptr, XNh);
    // 2) QKV projection (f16 out)
    gemm_wmma<EP_BIAS><<<dim3(3 * E_ / BN, T_ / BM), blk, 0, stream>>>(
        XNh, W_INh, in_b, QKVh, T_, 3 * E_, E_, nullptr, 0, 0, nullptr);
    // 3) Attention
    attn_kernel<<<T_, 512, 0, stream>>>(QKVh, CTXh, out + OAW);
    // 4) Output projection, fused residual: X1 = x + ctx @ Wo^T + bo
    gemm_wmma<EP_RES><<<dim3(E_ / BN, T_ / BM), blk, 0, stream>>>(
        CTXh, W_OUTh, out_b, X1, T_, E_, E_, nullptr, 0, 0, x);
    // 5) LN2 (fp32 for router, f16 for GEMMs)
    ln_kernel<<<lnBlocks, blk, 0, stream>>>(X1, norm2_w, norm2_b, XN2, XN2h);
    // 6) Routers
    zero_kernel<<<1, 32, 0, stream>>>(RS, 12);
    router_kernel<NS_><<<lnBlocks, blk, 0, stream>>>(XN2, shared_rw, shared_rb,
                                                     out + OSM, RS + 0, RS + NS_);
    router_kernel<NM_><<<lnBlocks, blk, 0, stream>>>(XN2, image_rw, image_rb,
                                                     out + OIM, RS + 8, RS + 10);
    // 7) CMB = X1 (experts accumulate on top)
    copy_kernel<<<(T_ * E_) / 256, blk, 0, stream>>>(X1, CMB, T_ * E_);
    // 8) Experts
    for (int n = 0; n < NE_; ++n) {
        gemm_wmma<EP_GELU><<<dim3(DFF_ / BN, T_ / BM), blk, 0, stream>>>(
            XN2h, W_E1h + (size_t)n * DFF_ * E_, exp_b1 + (size_t)n * DFF_,
            HBh, T_, DFF_, E_, nullptr, 0, 0, nullptr);
        const float* mk; int mld, mcol;
        if (n < NS_) { mk = out + OSM; mld = NS_; mcol = n; }
        else         { mk = out + OIM; mld = NM_; mcol = n - NS_; }
        gemm_wmma<EP_MASK><<<dim3(E_ / BN, T_ / BM), blk, 0, stream>>>(
            HBh, W_E2h + (size_t)n * E_ * DFF_, exp_b2 + (size_t)n * E_,
            CMB, T_, E_, DFF_, mk, mld, mcol, nullptr);
    }
    // 9) LN3 (reuses norm2 params)
    ln_kernel<<<lnBlocks, blk, 0, stream>>>(CMB, norm2_w, norm2_b, nullptr, XNh);
    // 10) FFN
    gemm_wmma<EP_GELU><<<dim3(DFF_ / BN, T_ / BM), blk, 0, stream>>>(
        XNh, W_F1h, ffn_b1, HBh, T_, DFF_, E_, nullptr, 0, 0, nullptr);
    gemm_wmma<EP_RES><<<dim3(E_ / BN, T_ / BM), blk, 0, stream>>>(
        HBh, W_F2h, ffn_b2, out + OX, T_, E_, DFF_, nullptr, 0, 0, CMB);
    // 11) Router balance loss
    loss_kernel<<<1, 32, 0, stream>>>(RS, out + OL);
}
